// pytorchLSTM_36747740184611
// MI455X (gfx1250) — matrix operand order repaired and checked by reading, not compiled, not run on hardware
//
#include <hip/hip_runtime.h>
#include <hip/hip_bf16.h>
#include <stdint.h>

typedef __attribute__((ext_vector_type(16))) __bf16 v16bf;
typedef __attribute__((ext_vector_type(8)))  float  v8f;

#define LN_EPS 1e-5f

// ---------------- utility kernels ----------------

__global__ void fill_zero_u32(uint32_t* __restrict__ p, int n) {
    int i = blockIdx.x * blockDim.x + threadIdx.x;
    if (i < n) p[i] = 0u;
}

__global__ void cvt_f32_bf16(const float* __restrict__ s, __bf16* __restrict__ d, int n) {
    int i = blockIdx.x * blockDim.x + threadIdx.x;
    int stride = gridDim.x * blockDim.x;
    for (; i < n; i += stride) d[i] = (__bf16)s[i];
}

// ---------------- WMMA GEMM: out[M,N] = A[M,K] @ W[N,K]^T + bias[N] ----------------
// A, W bf16 row-major; K % 32 == 0, M % (16*MT) == 0, N % (64*NT) == 0.
// Block = 128 threads (4 waves). Each wave computes a (16*MT) x (16*NT) register tile:
// MT A-fragments + NT B-fragments feed MT*NT WMMAs per K-step.
template <int MT, int NT, bool OUT_BF16>
__global__ void gemm_bf16_wmma(const __bf16* __restrict__ A,
                               const __bf16* __restrict__ W,
                               const float*  __restrict__ bias,
                               void* __restrict__ out,
                               int M, int N, int K)
{
    const int lane = threadIdx.x & 31;
    const int wave = threadIdx.x >> 5;
    const int n0   = (blockIdx.x * 4 + wave) * (16 * NT);
    const int m0   = blockIdx.y * (16 * MT);
    const int half = lane >> 4;   // 0: lanes 0-15, 1: lanes 16-31
    const int l15  = lane & 15;

    const __bf16* __restrict__ Arow[MT];
#pragma unroll
    for (int mi = 0; mi < MT; ++mi)
        Arow[mi] = A + (size_t)(m0 + mi * 16 + l15) * K;

    const __bf16* __restrict__ Wrow[NT];
    int colB[NT];
    v8f acc[MT][NT];
#pragma unroll
    for (int ni = 0; ni < NT; ++ni) {
        colB[ni] = n0 + ni * 16 + l15;
        Wrow[ni] = W + (size_t)colB[ni] * K;
        const float bv = bias[colB[ni]];
#pragma unroll
        for (int mi = 0; mi < MT; ++mi)
#pragma unroll
            for (int i = 0; i < 8; ++i) acc[mi][ni][i] = bv;
    }

    for (int k0 = 0; k0 < K; k0 += 32) {
        // A fragments (16x32 bf16): lane<16 holds K k0..k0+15,
        // lane>=16 holds K k0+16..k0+31
        union { v16bf v; uint4 u[2]; } af[MT];
        const int ka = k0 + half * 16;
#pragma unroll
        for (int mi = 0; mi < MT; ++mi) {
            af[mi].u[0] = *(const uint4*)(Arow[mi] + ka);
            af[mi].u[1] = *(const uint4*)(Arow[mi] + ka + 8);
        }
        // B fragments (32x16 bf16): lane<16 holds K k0..k0+15 of column colB,
        // lane>=16 holds K k0+16..k0+31 (W is [N,K] row-major -> contiguous)
        v16bf bfrag[NT];
#pragma unroll
        for (int ni = 0; ni < NT; ++ni)
            bfrag[ni] = *(const v16bf*)(Wrow[ni] + k0 + half * 16);

        if (k0 + 32 < K) {
            // prefetch next K-chunk of the streamed weight rows (global_prefetch_b8)
            __builtin_prefetch(Wrow[0] + k0 + 32, 0, 1);
            __builtin_prefetch(Arow[0] + k0 + 32, 0, 1);
        }

#pragma unroll
        for (int mi = 0; mi < MT; ++mi)
#pragma unroll
            for (int ni = 0; ni < NT; ++ni)
                acc[mi][ni] = __builtin_amdgcn_wmma_f32_16x16x32_bf16(
                    false, af[mi].v, false, bfrag[ni], (short)0, acc[mi][ni], false, false);
    }

    // C/D layout: vgpr i -> row m_tile_base + i + 8*half, column colB[ni]
#pragma unroll
    for (int mi = 0; mi < MT; ++mi) {
        const int rbase = m0 + mi * 16 + half * 8;
#pragma unroll
        for (int ni = 0; ni < NT; ++ni) {
            if (OUT_BF16) {
                __bf16* o = (__bf16*)out;
#pragma unroll
                for (int i = 0; i < 8; ++i)
                    o[(size_t)(rbase + i) * N + colB[ni]] = (__bf16)acc[mi][ni][i];
            } else {
                float* o = (float*)out;
#pragma unroll
                for (int i = 0; i < 8; ++i)
                    o[(size_t)(rbase + i) * N + colB[ni]] = acc[mi][ni][i];
            }
        }
    }
}

// ---------------- LayerNorm in-place over rows of bf16 [rows, 4096] ----------------
__global__ void ln_rows_bf16(__bf16* __restrict__ g,
                             const float* __restrict__ gamma,
                             const float* __restrict__ beta,
                             int Nc /* = 4096 */)
{
    const int row = blockIdx.x;
    const int tid = threadIdx.x;       // 256 threads, 16 elems each
    __bf16* p = g + (size_t)row * Nc;

    float v[16];
    float s = 0.f, sq = 0.f;
#pragma unroll
    for (int i = 0; i < 16; ++i) {
        float x = (float)p[tid * 16 + i];
        v[i] = x; s += x; sq += x * x;
    }
    __shared__ float rs[256], rq[256];
    rs[tid] = s; rq[tid] = sq; __syncthreads();
    for (int off = 128; off > 0; off >>= 1) {
        if (tid < off) { rs[tid] += rs[tid + off]; rq[tid] += rq[tid + off]; }
        __syncthreads();
    }
    const float inv = 1.f / (float)Nc;
    float mean = rs[0] * inv;
    float var  = rq[0] * inv - mean * mean;
    float rstd = rsqrtf(var + LN_EPS);
#pragma unroll
    for (int i = 0; i < 16; ++i) {
        int j = tid * 16 + i;
        p[j] = (__bf16)((v[i] - mean) * rstd * gamma[j] + beta[j]);
    }
}

// ---------------- fused LSTM cell (one block per batch row, 256 threads) ----------------
__global__ void lstm_cell(const float*  __restrict__ hgates,  // [64,4096] raw (+b_hh)
                          const __bf16* __restrict__ xgates,  // [B*T,4096] LN'd
                          const float* __restrict__ g_hh, const float* __restrict__ be_hh,
                          const float* __restrict__ g_ho, const float* __restrict__ be_ho,
                          float*  __restrict__ c,     // [64,1024] fp32 cell state
                          __bf16* __restrict__ h_bf,  // [64,1024] next-step h (bf16)
                          __bf16* __restrict__ y_bf,  // [B*T,1024] layer output
                          int t, int T)
{
    const int b   = blockIdx.x;
    const int tid = threadIdx.x;   // 256
    const int j0  = tid * 4;
    const size_t rowx = ((size_t)b * T + t) * 4096;
    const float* __restrict__ hr = hgates + (size_t)b * 4096;

    // ---- LN stats over the 4096 raw h-gates ----
    float hv[4][4];
    float s = 0.f, sq = 0.f;
#pragma unroll
    for (int gi = 0; gi < 4; ++gi)
#pragma unroll
        for (int jj = 0; jj < 4; ++jj) {
            float x = hr[gi * 1024 + j0 + jj];
            hv[gi][jj] = x; s += x; sq += x * x;
        }
    __shared__ float rs[256], rq[256];
    rs[tid] = s; rq[tid] = sq; __syncthreads();
    for (int off = 128; off > 0; off >>= 1) {
        if (tid < off) { rs[tid] += rs[tid + off]; rq[tid] += rq[tid + off]; }
        __syncthreads();
    }
    float mean = rs[0] * (1.f / 4096.f);
    float var  = rq[0] * (1.f / 4096.f) - mean * mean;
    float rstd = rsqrtf(var + LN_EPS);
    __syncthreads();   // rs/rq reused below

    // ---- gates, cell update ----
    float cy[4], oo[4];
    float s2 = 0.f, q2 = 0.f;
#pragma unroll
    for (int jj = 0; jj < 4; ++jj) {
        const int j = j0 + jj;
        float gate[4];
#pragma unroll
        for (int gi = 0; gi < 4; ++gi) {
            const int col = gi * 1024 + j;
            gate[gi] = (float)xgates[rowx + col]
                     + (hv[gi][jj] - mean) * rstd * g_hh[col] + be_hh[col];
        }
        float i_ = 1.f / (1.f + __expf(-gate[0]));
        float f_ = 1.f / (1.f + __expf(-gate[1]));
        float o_ = 1.f / (1.f + __expf(-gate[2]));
        float g_ = tanhf(gate[3]);
        float cv = f_ * c[(size_t)b * 1024 + j] + i_ * g_;
        c[(size_t)b * 1024 + j] = cv;
        cy[jj] = cv; oo[jj] = o_;
        s2 += cv; q2 += cv * cv;
    }

    // ---- LN stats over the 1024 new cell values ----
    rs[tid] = s2; rq[tid] = q2; __syncthreads();
    for (int off = 128; off > 0; off >>= 1) {
        if (tid < off) { rs[tid] += rs[tid + off]; rq[tid] += rq[tid + off]; }
        __syncthreads();
    }
    float m2 = rs[0] * (1.f / 1024.f);
    float v2 = rq[0] * (1.f / 1024.f) - m2 * m2;
    float r2 = rsqrtf(v2 + LN_EPS);

#pragma unroll
    for (int jj = 0; jj < 4; ++jj) {
        const int j = j0 + jj;
        float hy = oo[jj] * tanhf((cy[jj] - m2) * r2 * g_ho[j] + be_ho[j]);
        __bf16 hb = (__bf16)hy;
        h_bf[(size_t)b * 1024 + j] = hb;
        y_bf[((size_t)b * T + t) * 1024 + j] = hb;
    }
}

// ---------------- host orchestration ----------------

extern "C" void kernel_launch(void* const* d_in, const int* in_sizes, int n_in,
                              void* d_out, int out_size, void* d_ws, size_t ws_size,
                              hipStream_t stream)
{
    (void)in_sizes; (void)n_in; (void)out_size; (void)ws_size;
    const float* input = (const float*)d_in[0];   // [64,512,1024]
    const float* w_ih  = (const float*)d_in[1];   // [2,4096,1024]
    const float* b_ih  = (const float*)d_in[2];   // [2,4096]
    const float* w_hh  = (const float*)d_in[3];   // [2,4096,1024]
    const float* b_hh  = (const float*)d_in[4];   // [2,4096]
    const float* g_ih  = (const float*)d_in[5];   // [2,4096]
    const float* be_ih = (const float*)d_in[6];
    const float* g_hh  = (const float*)d_in[7];
    const float* be_hh = (const float*)d_in[8];
    const float* g_ho  = (const float*)d_in[9];   // [2,1024]
    const float* be_ho = (const float*)d_in[10];
    const float* fc_w  = (const float*)d_in[11];  // [1024,1024]
    const float* fc_b  = (const float*)d_in[12];  // [1024]

    const int B = 64, T = 512, D = 1024, H = 1024, G = 4096, depth = 2, C = 1024;
    const size_t BT = (size_t)B * T;   // 32768

    // ---- workspace partition (256B aligned) ----
    uint8_t* ws = (uint8_t*)d_ws;
    size_t off = 0;
    auto wsalloc = [&](size_t bytes) -> uint8_t* {
        uint8_t* p = ws + off;
        off += (bytes + 255) & ~(size_t)255;
        return p;
    };
    __bf16* wih_bf = (__bf16*)wsalloc((size_t)depth * G * D * 2);   // 16 MB
    __bf16* whh_bf = (__bf16*)wsalloc((size_t)depth * G * H * 2);   // 16 MB
    __bf16* fcw_bf = (__bf16*)wsalloc((size_t)C * H * 2);           //  2 MB
    __bf16* Xa     = (__bf16*)wsalloc(BT * (size_t)D * 2);          // 64 MB
    __bf16* Xb     = (__bf16*)wsalloc(BT * (size_t)H * 2);          // 64 MB
    __bf16* xg     = (__bf16*)wsalloc(BT * (size_t)G * 2);          // 256 MB
    float*  hg     = (float*) wsalloc((size_t)B * G * 4);           //  1 MB
    float*  cst    = (float*) wsalloc((size_t)B * H * 4);
    __bf16* hbf    = (__bf16*)wsalloc((size_t)B * H * 2);

    // ---- conversions (re-done every call; deterministic) ----
    cvt_f32_bf16<<<2048, 256, 0, stream>>>(w_ih, wih_bf, depth * G * D);
    cvt_f32_bf16<<<2048, 256, 0, stream>>>(w_hh, whh_bf, depth * G * H);
    cvt_f32_bf16<<<512,  256, 0, stream>>>(fc_w, fcw_bf, C * H);
    cvt_f32_bf16<<<4096, 256, 0, stream>>>(input, Xa, (int)(BT * D));

    __bf16* Xin = Xa;
    __bf16* Xout = Xb;

    for (int l = 0; l < depth; ++l) {
        // zero c (fp32) and h (bf16) state
        { int n = B * H;     fill_zero_u32<<<(n + 255) / 256, 256, 0, stream>>>((uint32_t*)cst, n); }
        { int n = B * H / 2; fill_zero_u32<<<(n + 255) / 256, 256, 0, stream>>>((uint32_t*)hbf, n); }

        // batched input-to-hidden GEMM for all T: xg = Xin @ Wih^T + b_ih (bf16 out)
        // fat tiles: wave = 32x64, block = 32x256 -> grid (16, 1024)
        {
            dim3 grid(G / 256, (unsigned)(BT / 32));
            gemm_bf16_wmma<2, 4, true><<<grid, 128, 0, stream>>>(
                Xin, wih_bf + (size_t)l * G * D, b_ih + (size_t)l * G,
                xg, (int)BT, G, D);
        }
        // LayerNorm(x-gates) in place with g_ih/be_ih
        ln_rows_bf16<<<(unsigned)BT, 256, 0, stream>>>(xg, g_ih + (size_t)l * G,
                                                       be_ih + (size_t)l * G, G);

        // sequential recurrence
        for (int t = 0; t < T; ++t) {
            // latency-bound small GEMM: wave = 16x32, block = 16x128 -> grid (32, 4) = 128 blocks
            dim3 grid(G / 128, B / 16);
            gemm_bf16_wmma<1, 2, false><<<grid, 128, 0, stream>>>(
                hbf, whh_bf + (size_t)l * G * H, b_hh + (size_t)l * G,
                hg, B, G, H);
            lstm_cell<<<B, 256, 0, stream>>>(
                hg, xg,
                g_hh + (size_t)l * G, be_hh + (size_t)l * G,
                g_ho + (size_t)l * H, be_ho + (size_t)l * H,
                cst, hbf, Xout, t, T);
        }
        // next layer consumes this layer's output
        __bf16* tmp = Xin; Xin = Xout; Xout = tmp;
    }

    // final FC: d_out[BT, C] = Y @ fc_w^T + fc_b  (fp32 out), fat tiles -> grid (4, 1024)
    {
        dim3 grid(C / 256, (unsigned)(BT / 32));
        gemm_bf16_wmma<2, 4, false><<<grid, 128, 0, stream>>>(
            Xin, fcw_bf, fc_b, d_out, (int)BT, C, H);
    }
}
